// SpMultiHeadHigherOrderAttention_19696720019804
// MI455X (gfx1250) — matrix operand order repaired and checked by reading, not compiled, not run on hardware
//
#include <hip/hip_runtime.h>

#define N_NODES 20000
#define N_EDGES 320000
#define IN_F    256
#define OUT_F   64
#define HEADS   4
#define NF      (HEADS * OUT_F)   // 256 = concat width
#define ALPHA   0.1f

typedef __attribute__((ext_vector_type(16))) _Float16 v16h;
typedef __attribute__((ext_vector_type(8)))  float    v8f;

__device__ __forceinline__ float lrelu(float x) { return x > 0.f ? x : ALPHA * x; }

// Order-preserving float<->uint encoding so atomicMax(u32) == float max
__device__ __forceinline__ unsigned fenc(float x) {
  unsigned u = __float_as_uint(x);
  return (u & 0x80000000u) ? ~u : (u | 0x80000000u);
}
__device__ __forceinline__ float fdec(unsigned u) {
  unsigned v = (u & 0x80000000u) ? (u & 0x7fffffffu) : ~u;
  return __uint_as_float(v);
}

// ---------------------------------------------------------------------------
// Kernel 1: S = X @ Wall  (X: [N_NODES, IN_F] f32, Wall: [HEADS, IN_F, OUT_F])
// One wave computes a 16x16 f32 tile via v_wmma_f32_16x16x32_f16; the K=256
// reduction is fully unrolled into 8 back-to-back WMMA issues so the compiler
// can clause the tile loads and overlap cvt/load with matrix issue.
// grid (1250, 4), block 128 (4 waves); wave w -> ntile = blockIdx.y*4 + w.
// ---------------------------------------------------------------------------
__global__ __launch_bounds__(128) void gemm_wmma_f16(
    const float* __restrict__ X, const float* __restrict__ Wall,
    float* __restrict__ Out) {
  const int lane = threadIdx.x & 31;
  const int wave = threadIdx.x >> 5;
  const int m0 = blockIdx.x * 16;
  const int ntile = blockIdx.y * 4 + wave;      // 0..15
  const int n0 = ntile * 16;
  const int half = lane >> 4;                   // 0: lanes 0-15, 1: lanes 16-31
  const int ml = lane & 15;                     // A: M index, B: N index

  const int h    = n0 >> 6;                     // head
  const int ocol = n0 & 63;                     // column base inside head
  const float* Wh = Wall + (size_t)h * (IN_F * OUT_F);

  const float* arow_base = X + (size_t)(m0 + ml) * IN_F;

  // prefetch next M-tile's A row into L2 (global_prefetch_b8)
  if (blockIdx.x + 1 < gridDim.x)
    __builtin_prefetch(arow_base + (size_t)16 * IN_F, 0, 1);

  v8f acc = {};
#pragma unroll
  for (int k0 = 0; k0 < IN_F; k0 += 32) {
    // ---- A tile: row (m0+ml); element i -> K = k0 + 16*half + i
    v16h a;
    const float* arow = arow_base + k0 + 16 * half;
#pragma unroll
    for (int i = 0; i < 8; ++i) a[i] = (_Float16)arow[i];
#pragma unroll
    for (int i = 0; i < 8; ++i) a[8 + i] = (_Float16)arow[8 + i];

    // ---- B tile: column n = ocol+ml; element i -> K = k0 + 16*half + i
    v16h b;
    const float* bcol = Wh + (size_t)(k0 + 16 * half) * OUT_F + (ocol + ml);
#pragma unroll
    for (int i = 0; i < 16; ++i) b[i] = (_Float16)bcol[(size_t)i * OUT_F];

    acc = __builtin_amdgcn_wmma_f32_16x16x32_f16(
        /*neg_a=*/false, a, /*neg_b=*/false, b,
        /*c_mod=*/(short)0, acc, /*reuse_a=*/false, /*reuse_b=*/false);
  }

  // D layout: VGPR r -> M = 8*half + r, N = ml
  float* orow = Out + (size_t)(m0 + 8 * half) * NF + n0 + ml;
#pragma unroll
  for (int r = 0; r < 8; ++r) orow[(size_t)r * NF] = acc[r];
}

// ---------------------------------------------------------------------------
// Kernel 2: per-(node,head) dots with attention vectors
// sa1 = S_h . a1[:64], ta1 = T_h . a1[64:], same for a2.  idx = node*4 + h
// ---------------------------------------------------------------------------
__global__ __launch_bounds__(256) void rowdot_kernel(
    const float* __restrict__ S, const float* __restrict__ T,
    const float* __restrict__ a1, const float* __restrict__ a2,
    float* __restrict__ sa1, float* __restrict__ ta1,
    float* __restrict__ sa2, float* __restrict__ ta2) {
  int idx = blockIdx.x * blockDim.x + threadIdx.x;
  if (idx >= N_NODES * HEADS) return;
  int n = idx >> 2, h = idx & 3;
  const float* srow = S + (size_t)n * NF + h * OUT_F;
  const float* trow = T + (size_t)n * NF + h * OUT_F;
  const float* a1h = a1 + h * (2 * OUT_F);
  const float* a2h = a2 + h * (2 * OUT_F);
  float s1 = 0.f, t1 = 0.f, s2 = 0.f, t2 = 0.f;
#pragma unroll 8
  for (int o = 0; o < OUT_F; ++o) {
    float sv = srow[o], tv = trow[o];
    s1 += sv * a1h[o];          t1 += tv * a1h[OUT_F + o];
    s2 += sv * a2h[o];          t2 += tv * a2h[OUT_F + o];
  }
  sa1[idx] = s1; ta1[idx] = t1; sa2[idx] = s2; ta2[idx] = t2;
}

// ---------------------------------------------------------------------------
// Kernel 3: init accumulators (d_out, segment sums, segment maxes)
// ---------------------------------------------------------------------------
__global__ __launch_bounds__(256) void init_kernel(
    float* __restrict__ out, unsigned* __restrict__ m1, unsigned* __restrict__ m2,
    float* __restrict__ d1, float* __restrict__ d2) {
  int idx = blockIdx.x * blockDim.x + threadIdx.x;
  if (idx < 2 * N_NODES * NF) out[idx] = 0.f;
  if (idx < N_NODES * HEADS) {
    unsigned ninf = fenc(-__builtin_inff());
    m1[idx] = ninf; m2[idx] = ninf;
    d1[idx] = 0.f;  d2[idx] = 0.f;
  }
}

// ---------------------------------------------------------------------------
// Kernel 4: segment max over edges (per edge, all 4 heads)
// ---------------------------------------------------------------------------
__global__ __launch_bounds__(256) void edge_max_kernel(
    const int* __restrict__ src, const int* __restrict__ tgt,
    const float* __restrict__ sa1, const float* __restrict__ ta1,
    const float* __restrict__ sa2, const float* __restrict__ ta2,
    unsigned* __restrict__ m1, unsigned* __restrict__ m2) {
  int e = blockIdx.x * blockDim.x + threadIdx.x;
  if (e >= N_EDGES) return;
  int s = src[e], t = tgt[e];
#pragma unroll
  for (int h = 0; h < HEADS; ++h) {
    float e1 = lrelu(sa1[4 * s + h] + ta1[4 * t + h]);
    float e2 = lrelu(sa2[4 * s + h] + ta2[4 * t + h]);
    atomicMax(&m1[4 * t + h], fenc(e1));   // softmax grouped by tgt
    atomicMax(&m2[4 * s + h], fenc(e2));   // softmax grouped by src
  }
}

// ---------------------------------------------------------------------------
// Kernel 5: w = exp(e - max), segment sums; store w per (edge, head)
// ---------------------------------------------------------------------------
__global__ __launch_bounds__(256) void edge_sum_kernel(
    const int* __restrict__ src, const int* __restrict__ tgt,
    const float* __restrict__ sa1, const float* __restrict__ ta1,
    const float* __restrict__ sa2, const float* __restrict__ ta2,
    const unsigned* __restrict__ m1, const unsigned* __restrict__ m2,
    float* __restrict__ d1, float* __restrict__ d2,
    float* __restrict__ w1arr, float* __restrict__ w2arr) {
  int e = blockIdx.x * blockDim.x + threadIdx.x;
  if (e >= N_EDGES) return;
  int s = src[e], t = tgt[e];
#pragma unroll
  for (int h = 0; h < HEADS; ++h) {
    float e1 = lrelu(sa1[4 * s + h] + ta1[4 * t + h]);
    float e2 = lrelu(sa2[4 * s + h] + ta2[4 * t + h]);
    float w1 = __expf(e1 - fdec(m1[4 * t + h]));
    float w2 = __expf(e2 - fdec(m2[4 * s + h]));
    w1arr[4 * e + h] = w1;  atomicAdd(&d1[4 * t + h], w1);
    w2arr[4 * e + h] = w2;  atomicAdd(&d2[4 * s + h], w2);
  }
}

// ---------------------------------------------------------------------------
// Kernel 6: scatter. One wave32 per (edge, head); lane owns 2 features.
// h_ts[tgt] += (w1/d1[tgt]) * S[src];  h_st[src] += (w2/d2[src]) * T[tgt]
// Whole working set is L2-resident (192 MB), so atomics run at L2 rate.
// ---------------------------------------------------------------------------
__global__ __launch_bounds__(256) void edge_scatter_kernel(
    const int* __restrict__ src, const int* __restrict__ tgt,
    const float* __restrict__ S, const float* __restrict__ T,
    const float* __restrict__ w1arr, const float* __restrict__ w2arr,
    const float* __restrict__ d1, const float* __restrict__ d2,
    float* __restrict__ out_st, float* __restrict__ out_ts) {
  int gw = (blockIdx.x * blockDim.x + threadIdx.x) >> 5;  // (edge,head) id
  int lane = threadIdx.x & 31;
  if (gw >= N_EDGES * HEADS) return;
  int e = gw >> 2, h = gw & 3;
  int s = src[e], t = tgt[e];
  float c1 = w1arr[gw] / d1[4 * t + h];   // d > 0: segment has >=1 edge
  float c2 = w2arr[gw] / d2[4 * s + h];
  int o = 2 * lane;
  const float2 sv = *(const float2*)(S + (size_t)s * NF + h * OUT_F + o);
  const float2 tv = *(const float2*)(T + (size_t)t * NF + h * OUT_F + o);
  float* pts = out_ts + (size_t)t * NF + h * OUT_F + o;
  float* pst = out_st + (size_t)s * NF + h * OUT_F + o;
  atomicAdd(pts,     c1 * sv.x);
  atomicAdd(pts + 1, c1 * sv.y);
  atomicAdd(pst,     c2 * tv.x);
  atomicAdd(pst + 1, c2 * tv.y);
}

// ---------------------------------------------------------------------------
// Kernel 7: ELU in place over d_out (both halves)
// ---------------------------------------------------------------------------
__global__ __launch_bounds__(256) void elu_kernel(float* __restrict__ out, int n) {
  int i = blockIdx.x * blockDim.x + threadIdx.x;
  if (i < n) {
    float x = out[i];
    out[i] = x > 0.f ? x : (__expf(x) - 1.0f);
  }
}

// ---------------------------------------------------------------------------
extern "C" void kernel_launch(void* const* d_in, const int* in_sizes, int n_in,
                              void* d_out, int out_size, void* d_ws, size_t ws_size,
                              hipStream_t stream) {
  const float* input1 = (const float*)d_in[0];
  const float* input2 = (const float*)d_in[1];
  const float* Ws     = (const float*)d_in[2];
  const float* Wt     = (const float*)d_in[3];
  const float* a1     = (const float*)d_in[4];
  const float* a2     = (const float*)d_in[5];
  const int*   tgt    = (const int*)d_in[6];
  const int*   src    = (const int*)d_in[7];

  float* out_st = (float*)d_out;                          // h_st [20000,256]
  float* out_ts = out_st + (size_t)N_NODES * NF;          // h_ts [20000,256]

  // workspace layout (floats)
  float* ws = (float*)d_ws;
  float* S     = ws;                         // 5,120,000
  float* T     = S    + (size_t)N_NODES * NF;
  float* sa1   = T    + (size_t)N_NODES * NF;     //   80,000 each
  float* ta1   = sa1  + (size_t)N_NODES * HEADS;
  float* sa2   = ta1  + (size_t)N_NODES * HEADS;
  float* ta2   = sa2  + (size_t)N_NODES * HEADS;
  float* d1    = ta2  + (size_t)N_NODES * HEADS;
  float* d2    = d1   + (size_t)N_NODES * HEADS;
  float* w1arr = d2   + (size_t)N_NODES * HEADS;  // 1,280,000 each
  float* w2arr = w1arr + (size_t)N_EDGES * HEADS;
  unsigned* m1 = (unsigned*)(w2arr + (size_t)N_EDGES * HEADS);
  unsigned* m2 = m1 + (size_t)N_NODES * HEADS;

  // 1) WMMA GEMMs: S = x1 @ Ws_all, T = x2 @ Wt_all
  dim3 ggrid(N_NODES / 16, 4, 1);
  gemm_wmma_f16<<<ggrid, 128, 0, stream>>>(input1, Ws, S);
  gemm_wmma_f16<<<ggrid, 128, 0, stream>>>(input2, Wt, T);

  // 2) attention-vector dots
  rowdot_kernel<<<(N_NODES * HEADS + 255) / 256, 256, 0, stream>>>(
      S, T, a1, a2, sa1, ta1, sa2, ta2);

  // 3) init accumulators (also zeros d_out, which we accumulate into)
  init_kernel<<<(2 * N_NODES * NF + 255) / 256, 256, 0, stream>>>(
      out_st, m1, m2, d1, d2);

  // 4) segment max
  edge_max_kernel<<<(N_EDGES + 255) / 256, 256, 0, stream>>>(
      src, tgt, sa1, ta1, sa2, ta2, m1, m2);

  // 5) exp + segment sum
  edge_sum_kernel<<<(N_EDGES + 255) / 256, 256, 0, stream>>>(
      src, tgt, sa1, ta1, sa2, ta2, m1, m2, d1, d2, w1arr, w2arr);

  // 6) scatter: one wave per (edge, head)
  {
    long long waves = (long long)N_EDGES * HEADS;          // 1,280,000
    long long threads = waves * 32;
    int blocks = (int)((threads + 255) / 256);             // 160,000
    edge_scatter_kernel<<<blocks, 256, 0, stream>>>(
        src, tgt, S, T, w1arr, w2arr, d1, d2, out_st, out_ts);
  }

  // 7) ELU in place
  elu_kernel<<<(2 * N_NODES * NF + 255) / 256, 256, 0, stream>>>(
      out_st, 2 * N_NODES * NF);
}
